// MultiHeadAttention_30992484008530
// MI455X (gfx1250) — matrix operand order repaired and checked by reading, not compiled, not run on hardware
//
#include <hip/hip_runtime.h>
#include <hip/hip_bf16.h>
#include <math.h>

typedef __attribute__((ext_vector_type(16))) _Float16 v16h;
typedef __attribute__((ext_vector_type(8)))  float    v8f;
typedef __attribute__((ext_vector_type(4)))  unsigned int u32x4;
typedef __attribute__((ext_vector_type(8)))  int i32x8;
typedef __attribute__((ext_vector_type(4)))  int i32x4;

#define WMMA_F16(A, B, C) \
  __builtin_amdgcn_wmma_f32_16x16x32_f16(false, (A), false, (B), (short)0, (C), false, false)

// Problem constants
#define BATCH 4
#define SEQ   2048
#define DMODEL 1024
#define NHEAD 16
#define DHEAD 64
#define NEG_BIG (-4294967295.0f)   // -2^32 + 1

// ws layout (f16 elements): Q [B][H][S][64] | K [B][H][S][64] | V^T [B][H][64][S]
#define QKV_ELEMS (BATCH * NHEAD * SEQ * DHEAD)   // 8388608

#if defined(__has_builtin)
# if __has_builtin(__builtin_amdgcn_tensor_load_to_lds) && \
     __has_builtin(__builtin_amdgcn_s_wait_tensorcnt)
#  define HAVE_TDM 1
# endif
#endif
#ifndef HAVE_TDM
# define HAVE_TDM 0
#endif

// --- CDNA5 async global->LDS copy (ASYNCcnt path, fallback) -----------------
__device__ __forceinline__ void async_copy_b128(void* lds_dst, const void* gsrc) {
  uint32_t loff = (uint32_t)(uintptr_t)lds_dst;   // flat->LDS = low 32 bits
  uint64_t ga   = (uint64_t)(uintptr_t)gsrc;
  asm volatile("global_load_async_to_lds_b128 %0, %1, off"
               :: "v"(loff), "v"(ga) : "memory");
}
__device__ __forceinline__ void wait_async_all() {
  asm volatile("s_wait_asynccnt 0x0" ::: "memory");
}

#if HAVE_TDM
// --- CDNA5 Tensor Data Mover: 2D tile global->LDS (TENSORcnt path) ----------
// D# per ISA 8.3/8.4: group0 = {count=1 | lds_addr | global_addr | type=2},
// group1 = {data_size=4B, tensor_dim0/1, tile_dim0/1, tensor_dim0_stride},
// groups 2/3 zero (<=2D tensor). All fields uniform per block -> SGPRs.
// This toolchain exposes the 6-arg builtin (extra int32x8 group before cpol).
__device__ __forceinline__ void tdm_load_2d(uint32_t lds_off, const void* gaddr,
                                            uint32_t tensor_w_dw,  // tensor_dim0 (dwords)
                                            uint32_t tensor_h,     // tensor_dim1 (rows)
                                            uint32_t stride_dw,    // tensor_dim0_stride
                                            uint32_t tile_w_dw,    // tile_dim0 (dwords)
                                            uint32_t tile_h)       // tile_dim1 (rows)
{
  const uint64_t ga = (uint64_t)(uintptr_t)gaddr;
  u32x4 g0;
  g0[0] = 1u;                                   // count=1 (valid), user mode
  g0[1] = lds_off;                              // lds_addr (bytes)
  g0[2] = (uint32_t)ga;                         // global_addr[31:0]
  g0[3] = (uint32_t)(ga >> 32) | (2u << 30);    // global_addr[56:32] | type=2
  i32x8 g1;
  g1[0] = (int)(2u << 16);                      // wg_mask=0, data_size=2 (4B)
  g1[1] = (int)(tensor_w_dw << 16);             // [63:48] = tensor_dim0 lo16
  g1[2] = (int)((tensor_w_dw >> 16) | (tensor_h << 16));  // dim0 hi16 | dim1 lo16
  g1[3] = (int)((tensor_h >> 16) | (tile_w_dw << 16));    // dim1 hi16 | tile_dim0
  g1[4] = (int)(tile_h & 0xffffu);              // tile_dim1 | tile_dim2=0
  g1[5] = (int)stride_dw;                       // tensor_dim0_stride lo32
  g1[6] = 0;                                    // stride hi16 | dim1_stride lo16
  g1[7] = 0;
  const i32x4 gz4 = {0, 0, 0, 0};
  const i32x8 gz8 = {0, 0, 0, 0, 0, 0, 0, 0};
  __builtin_amdgcn_tensor_load_to_lds(g0, g1, gz4, gz4, gz8, 0);
}
#endif

// ---------------------------------------------------------------------------
// Kernel 1: fused QKV projection  Y = X*W + b  (f32 in, f16 out, head-major)
// grid: (N/64=16, M/128=64, 3); block: 256 (8 waves); wave = 32x32 patch.
// X tile (128x32 f32, 16KB) and W tile (32x64 f32, 8KB) staged via TDM/async.
// ---------------------------------------------------------------------------
__global__ __launch_bounds__(256)
void qkv_proj_kernel(const float* __restrict__ q, const float* __restrict__ k,
                     const float* __restrict__ v,
                     const float* __restrict__ Wq, const float* __restrict__ bq,
                     const float* __restrict__ Wk, const float* __restrict__ bk,
                     const float* __restrict__ Wv, const float* __restrict__ bv,
                     _Float16* __restrict__ ws)
{
  __shared__ __align__(16) float ldsX[128 * 32];   // [row][k]   16 KB
  __shared__ __align__(16) float ldsW[32 * 64];    // [k][col]    8 KB

  const int proj = blockIdx.z;
  const float* X;  const float* W;  const float* bias;  _Float16* Y;
  if (proj == 0)      { X = q; W = Wq; bias = bq; Y = ws; }
  else if (proj == 1) { X = k; W = Wk; bias = bk; Y = ws + QKV_ELEMS; }
  else                { X = v; W = Wv; bias = bv; Y = ws + 2 * QKV_ELEMS; }

  const int tid  = threadIdx.x;
  const int lane = tid & 31;
  const int wave = tid >> 5;
  const int lh = lane >> 4;            // half-wave (0/1)
  const int ll = lane & 15;
  const int wm = wave >> 1;            // 0..3
  const int wn = wave & 1;             // 0..1
  const int mblock = blockIdx.y * 128;
  const int nblock = blockIdx.x * 64;

  v8f acc[2][2] = {};

  for (int k0 = 0; k0 < DMODEL; k0 += 32) {
    __syncthreads();                   // previous tile fully consumed
#if HAVE_TDM
    if (tid < 32) {                    // one wave drives the TDM
      tdm_load_2d((uint32_t)(uintptr_t)&ldsX[0],
                  X + (size_t)mblock * DMODEL + k0,
                  DMODEL, BATCH * SEQ, DMODEL, /*tile*/ 32, 128);
      tdm_load_2d((uint32_t)(uintptr_t)&ldsW[0],
                  W + (size_t)k0 * DMODEL + nblock,
                  DMODEL, DMODEL, DMODEL, /*tile*/ 64, 32);
      __builtin_amdgcn_s_wait_tensorcnt((short)0);
    }
#else
    // X tile: 128 rows x 32 f32 = 1024 x 16B segments, 4 per thread
#pragma unroll
    for (int i = 0; i < 4; ++i) {
      const int seg = tid + i * 256;
      const int row = seg >> 3, c = (seg & 7) * 4;
      async_copy_b128(&ldsX[row * 32 + c],
                      X + (size_t)(mblock + row) * DMODEL + k0 + c);
    }
    // W tile: 32 rows x 64 f32 = 512 x 16B segments, 2 per thread
#pragma unroll
    for (int i = 0; i < 2; ++i) {
      const int seg = tid + i * 256;
      const int row = seg >> 4, c = (seg & 15) * 4;
      async_copy_b128(&ldsW[row * 64 + c],
                      W + (size_t)(k0 + row) * DMODEL + nblock + c);
    }
    wait_async_all();
#endif
    __syncthreads();

    // A fragments from LDS: lane=row, K groups per ISA layout (32B runs)
    v16h afr[2];
#pragma unroll
    for (int mt = 0; mt < 2; ++mt) {
      const float* xp = &ldsX[(wm * 32 + mt * 16 + ll) * 32 + lh * 8];
#pragma unroll
      for (int j = 0; j < 8; ++j) {
        afr[mt][j]     = (_Float16)xp[j];        // K = k0 + lh*8 + j
        afr[mt][8 + j] = (_Float16)xp[16 + j];   // K = k0 + 16 + lh*8 + j
      }
    }
    // B fragments from LDS: lane=col, K groups per ISA layout (lane-consecutive banks)
    v16h bfr[2];
#pragma unroll
    for (int nt = 0; nt < 2; ++nt) {
      const float* wp = &ldsW[(lh * 8) * 64 + wn * 32 + nt * 16 + ll];
#pragma unroll
      for (int j = 0; j < 16; ++j) bfr[nt][j] = (_Float16)wp[((j & 7) + (j >> 3) * 16) * 64];
    }
#pragma unroll
    for (int mt = 0; mt < 2; ++mt)
#pragma unroll
      for (int nt = 0; nt < 2; ++nt)
        acc[mt][nt] = WMMA_F16(afr[mt], bfr[nt], acc[mt][nt]);
  }

  // bias + store in head-major layout (V stored transposed for attention)
#pragma unroll
  for (int nt = 0; nt < 2; ++nt) {
    const int c = nblock + wn * 32 + nt * 16 + ll;     // model column
    const float bval = bias[c];
    const int hh = c >> 6, dd = c & 63;
#pragma unroll
    for (int mt = 0; mt < 2; ++mt) {
#pragma unroll
      for (int j = 0; j < 8; ++j) {
        const int r  = mblock + wm * 32 + mt * 16 + j + lh * 8;  // C-layout row
        const int bb = r >> 11, ss = r & (SEQ - 1);
        const float val = acc[mt][nt][j] + bval;
        size_t addr;
        if (proj == 2) addr = ((size_t)((bb * NHEAD + hh) * DHEAD + dd)) * SEQ + ss;
        else           addr = ((size_t)((bb * NHEAD + hh)) * SEQ + ss) * DHEAD + dd;
        Y[addr] = (_Float16)val;
      }
    }
  }
}

// ---------------------------------------------------------------------------
// Kernel 2: flash attention. 8 waves/block share one (b,h): K/V chunks staged
// once per block (TDM or async LDS), one wave per 16-query tile, 32 keys per
// step. grid: (S/16/8=16, H=16, B=4). Output f32 [B][S][D].
// ---------------------------------------------------------------------------
__global__ __launch_bounds__(256)
void attn_kernel(const int* __restrict__ mask, const _Float16* __restrict__ ws,
                 float* __restrict__ out)
{
  const _Float16* Qh = ws;
  const _Float16* Kh = ws + QKV_ELEMS;
  const _Float16* Vt = ws + 2 * QKV_ELEMS;

  __shared__ __align__(16) _Float16 ldsK[32 * 64];       // [key][feat]  4 KB
  __shared__ __align__(16) _Float16 ldsV[64 * 32];       // [feat][key]  4 KB
  __shared__ __align__(16) _Float16 ldsP[8 * 16 * 32];   // per-wave P   8 KB

  const int tid  = threadIdx.x;
  const int lane = tid & 31;
  const int wave = tid >> 5;
  const int lh = lane >> 4, ll = lane & 15;
  const int hh = blockIdx.y, bb = blockIdx.z;
  const int qbase = (blockIdx.x * 8 + wave) * 16;

  const _Float16* Qp = Qh + ((size_t)(bb * NHEAD + hh)) * SEQ * DHEAD;
  const _Float16* Kp = Kh + ((size_t)(bb * NHEAD + hh)) * SEQ * DHEAD;
  const _Float16* Vp = Vt + ((size_t)(bb * NHEAD + hh)) * DHEAD * SEQ;
  _Float16* myP = ldsP + wave * (16 * 32);

  // Q A-fragments (features 0..31 and 32..63), loaded once
  v16h aq0, aq1;
  {
    const _Float16* qp = Qp + (size_t)(qbase + ll) * DHEAD + lh * 8;
#pragma unroll
    for (int j = 0; j < 8; ++j) {
      aq0[j]     = qp[j];        aq0[8 + j] = qp[16 + j];
      aq1[j]     = qp[32 + j];   aq1[8 + j] = qp[48 + j];
    }
  }

  // per-row online-softmax state; element j <-> row (j + 8*lh) (C layout)
  float m_i[8], l_i[8];
  int rmask[8];
#pragma unroll
  for (int j = 0; j < 8; ++j) {
    const int r = qbase + j + lh * 8;
    rmask[j] = mask[bb * SEQ + r];
    m_i[j] = -INFINITY;
    l_i[j] = 0.0f;
  }
  v8f acc[4] = {};

  for (int kt = 0; kt < SEQ; kt += 32) {
    __syncthreads();                    // previous chunk fully consumed
#if HAVE_TDM
    if (tid < 32) {                     // one wave drives the TDM
      // K chunk: 32 rows x 32 dwords (64 f16) out of [S x 32dw] tensor
      tdm_load_2d((uint32_t)(uintptr_t)&ldsK[0], Kp + (size_t)kt * DHEAD,
                  DHEAD / 2, SEQ, DHEAD / 2, /*tile*/ DHEAD / 2, 32);
      // V chunk (feature-major): 64 rows x 16 dwords (32 f16) of [64 x S/2dw]
      tdm_load_2d((uint32_t)(uintptr_t)&ldsV[0], Vp + kt,
                  SEQ / 2, DHEAD, SEQ / 2, /*tile*/ 16, DHEAD);
      __builtin_amdgcn_s_wait_tensorcnt((short)0);
    }
#else
    async_copy_b128(&ldsK[(tid >> 3) * 64 + (tid & 7) * 8],
                    Kp + (size_t)(kt + (tid >> 3)) * DHEAD + (tid & 7) * 8);
    async_copy_b128(&ldsV[(tid >> 2) * 32 + (tid & 3) * 8],
                    Vp + (size_t)(tid >> 2) * SEQ + kt + (tid & 3) * 8);
    wait_async_all();
#endif
    __syncthreads();

    // K^T B-fragments from LDS: [keytile 0/1][feature half 0/1]
    v16h bk[2][2];
#pragma unroll
    for (int t = 0; t < 2; ++t) {
      const _Float16* kp = &ldsK[(t * 16 + ll) * 64 + lh * 8];
#pragma unroll
      for (int j = 0; j < 16; ++j) { bk[t][0][j] = kp[(j & 7) + (j >> 3) * 16]; bk[t][1][j] = kp[32 + (j & 7) + (j >> 3) * 16]; }
    }
    v8f s0 = {}, s1 = {};
    s0 = WMMA_F16(aq0, bk[0][0], s0);  s0 = WMMA_F16(aq1, bk[0][1], s0);
    s1 = WMMA_F16(aq0, bk[1][0], s1);  s1 = WMMA_F16(aq1, bk[1][1], s1);

    // online softmax update (scale = 1/sqrt(64) = 0.125)
    float p0[8], p1[8];
#pragma unroll
    for (int j = 0; j < 8; ++j) {
      float v0 = s0[j] * 0.125f, v1 = s1[j] * 0.125f;
      if (rmask[j] == 0) { v0 = NEG_BIG; v1 = NEG_BIG; }
      float t = fmaxf(v0, v1);
#pragma unroll
      for (int off = 1; off < 16; off <<= 1) t = fmaxf(t, __shfl_xor(t, off, 32));
      const float mn = fmaxf(m_i[j], t);
      p0[j] = __expf(v0 - mn);
      p1[j] = __expf(v1 - mn);
      float psum = p0[j] + p1[j];
#pragma unroll
      for (int off = 1; off < 16; off <<= 1) psum += __shfl_xor(psum, off, 32);
      const float corr = __expf(m_i[j] - mn);
      l_i[j] = l_i[j] * corr + psum;
      m_i[j] = mn;
#pragma unroll
      for (int nt = 0; nt < 4; ++nt) acc[nt][j] *= corr;
    }

    // C-layout -> A-layout reshape of P through wave-private LDS (in-order DS)
#pragma unroll
    for (int j = 0; j < 8; ++j) {
      const int r = j + lh * 8;
      myP[r * 32 + ll]      = (_Float16)p0[j];
      myP[r * 32 + 16 + ll] = (_Float16)p1[j];
    }
    asm volatile("s_wait_dscnt 0x0" ::: "memory");
    v16h aP;
    {
      const _Float16* pp = myP + ll * 32 + lh * 8;
#pragma unroll
      for (int j = 0; j < 8; ++j) { aP[j] = pp[j]; aP[8 + j] = pp[16 + j]; }
    }

    // V B-fragments from LDS (K groups per ISA layout), P·V
#pragma unroll
    for (int nt = 0; nt < 4; ++nt) {
      v16h bv;
      const _Float16* vp = &ldsV[(nt * 16 + ll) * 32 + lh * 8];
#pragma unroll
      for (int j = 0; j < 16; ++j) bv[j] = vp[(j & 7) + (j >> 3) * 16];
      acc[nt] = WMMA_F16(aP, bv, acc[nt]);
    }
  }

  // normalize and store f32 output [B][S][D]
#pragma unroll
  for (int nt = 0; nt < 4; ++nt) {
    const int col = hh * DHEAD + nt * 16 + ll;
#pragma unroll
    for (int j = 0; j < 8; ++j) {
      const int r = qbase + j + lh * 8;
      out[((size_t)(bb * SEQ + r)) * DMODEL + col] = acc[nt][j] / l_i[j];
    }
  }
}

// ---------------------------------------------------------------------------
extern "C" void kernel_launch(void* const* d_in, const int* in_sizes, int n_in,
                              void* d_out, int out_size, void* d_ws, size_t ws_size,
                              hipStream_t stream) {
  (void)in_sizes; (void)n_in; (void)out_size; (void)ws_size;
  const float* q    = (const float*)d_in[0];
  const float* k    = (const float*)d_in[1];
  const float* v    = (const float*)d_in[2];
  const int*   mask = (const int*)  d_in[3];
  const float* Wq   = (const float*)d_in[4];
  const float* bq   = (const float*)d_in[5];
  const float* Wk   = (const float*)d_in[6];
  const float* bk   = (const float*)d_in[7];
  const float* Wv   = (const float*)d_in[8];
  const float* bv   = (const float*)d_in[9];
  _Float16* ws = (_Float16*)d_ws;
  float* out = (float*)d_out;

  // 3 projections: M=8192 (tile 128), N=1024 (tile 64)
  dim3 pgrid(DMODEL / 64, (BATCH * SEQ) / 128, 3);
  qkv_proj_kernel<<<pgrid, dim3(256), 0, stream>>>(q, k, v, Wq, bq, Wk, bk, Wv, bv, ws);

  // attention: 128 q-tiles / 8 waves per block
  dim3 agrid(SEQ / 16 / 8, NHEAD, BATCH);
  attn_kernel<<<agrid, dim3(256), 0, stream>>>(mask, ws, out);
}
